// Model_75582834475458
// MI455X (gfx1250) — matrix operand order repaired and checked by reading, not compiled, not run on hardware
//
#include <hip/hip_runtime.h>
#include <math.h>

// ---- problem constants (from reference) ----
#define BB   32
#define RR   5
#define CC   100
#define NN   101      // C+1
#define HH   128
#define EE   5
#define TT   5
#define NTIL 7        // ceil(101/16)
#define NEG_BIG 1e10f

typedef __attribute__((ext_vector_type(16))) _Float16 v16h;
typedef __attribute__((ext_vector_type(8)))  _Float16 v8h;
typedef __attribute__((ext_vector_type(2)))  _Float16 h2v;
typedef __attribute__((ext_vector_type(8)))  float    v8f;
typedef __attribute__((ext_vector_type(2)))  float    f2v;

// ---------------------------------------------------------------------------
// zero-fill
__global__ void k_zero(float* __restrict__ p, int n) {
    int i = blockIdx.x * blockDim.x + threadIdx.x;
    if (i < n) p[i] = 0.0f;
}

// ---------------------------------------------------------------------------
// Presence head: per (b,c) block computes h2 over all n, masked softmax over n,
// writes presence[b][n][c] (already transposed for later streaming).
__global__ void k_presence(const float* __restrict__ edge,
                           const float* __restrict__ avail,
                           const float* __restrict__ w1p,
                           const float* __restrict__ b1p,
                           const float* __restrict__ w2p,
                           const float* __restrict__ b2p,
                           float* __restrict__ presence) {
    __shared__ float w1l[EE * HH];
    __shared__ float b1l[HH];
    __shared__ float w2l[HH];
    __shared__ float sm[128];

    int bc = blockIdx.x;
    int b = bc / CC, c = bc % CC;
    int t = threadIdx.x;

    for (int i = t; i < EE * HH; i += 128) w1l[i] = w1p[i];
    b1l[t] = b1p[t];
    w2l[t] = w2p[t];
    __syncthreads();

    float logit = -3.0e38f;
    float ex = 0.0f;
    if (t < NN) {
        int n = t;
        const float* ep = edge + ((size_t)(b * CC + c) * NN + n) * EE;
        float e0 = ep[0], e1 = ep[1], e2 = ep[2], e3 = ep[3], e4 = ep[4];
        float acc = b2p[0];
        for (int h = 0; h < HH; ++h) {
            float a = b1l[h] + e0 * w1l[0 * HH + h] + e1 * w1l[1 * HH + h]
                    + e2 * w1l[2 * HH + h] + e3 * w1l[3 * HH + h] + e4 * w1l[4 * HH + h];
            a = fmaxf(a, 0.0f);
            acc += a * w2l[h];
        }
        float h2 = acc;                              // TAU == 1.0
        float m = (n == c || n == NN - 1) ? 0.0f : avail[b * NN + n];
        logit = h2 * m - (1.0f - m) * NEG_BIG;
    }
    sm[t] = logit; __syncthreads();
    for (int off = 64; off > 0; off >>= 1) {
        if (t < off) sm[t] = fmaxf(sm[t], sm[t + off]);
        __syncthreads();
    }
    float mx = sm[0]; __syncthreads();
    if (t < NN) ex = __expf(logit - mx);
    sm[t] = ex; __syncthreads();
    for (int off = 64; off > 0; off >>= 1) {
        if (t < off) sm[t] += sm[t + off];
        __syncthreads();
    }
    float inv = 1.0f / sm[0];
    if (t < NN) {
        float po = avail[b * NN + c] * ex * inv;     // avail[:, :, :-1] factor
        presence[((size_t)b * NN + t) * CC + c] = po;
    }
}

// ---------------------------------------------------------------------------
// fx1 = x @ wx1 + bx1, with x = [xa(5) | x_b(5) | coord(2) | avail(1)].
__global__ void k_fx1(const float* __restrict__ ap, const float* __restrict__ ac,
                      const float* __restrict__ xain, const float* __restrict__ xb,
                      const float* __restrict__ coord, const float* __restrict__ avail,
                      const float* __restrict__ wx1, const float* __restrict__ bx1,
                      float* __restrict__ fx) {
    int bn = blockIdx.x;
    int b = bn / NN, n = bn % NN;
    int t = threadIdx.x;

    float x[13];
    for (int f = 0; f < RR; ++f) {
        float s = 0.0f;
        for (int r = 0; r < RR; ++r) {
            int ai = (b * RR + r) * NN + n;
            float asg = ap[ai] + ac[ai];
            s += asg * xain[(size_t)ai * RR + f];
        }
        x[f] = s;
    }
    for (int f = 0; f < 5; ++f) x[5 + f] = xb[((size_t)b * NN + n) * 5 + f];
    x[10] = coord[((size_t)b * NN + n) * 2 + 0];
    x[11] = coord[((size_t)b * NN + n) * 2 + 1];
    x[12] = avail[b * NN + n];

    float acc = bx1[t];
    for (int k = 0; k < 13; ++k) acc += x[k] * wx1[k * HH + t];
    fx[((size_t)b * NN + n) * HH + t] = acc;
}

// ---------------------------------------------------------------------------
// Materialize pemb = presence * tanh(edge_dist @ we + be) in f16 (L2-resident, 82.7MB).
// grid = B*N blocks, 64 threads, each handles h pair (2t, 2t+1); 32-bit stores.
__global__ void k_pemb(const float* __restrict__ edge, const float* __restrict__ pres,
                       const float* __restrict__ we, const float* __restrict__ be,
                       _Float16* __restrict__ pemb) {
    int bn = blockIdx.x;
    int b = bn / NN, n = bn % NN;
    int h0 = 2 * threadIdx.x;
    f2v w0 = *(const f2v*)(we + 0 * HH + h0);
    f2v w1 = *(const f2v*)(we + 1 * HH + h0);
    f2v w2 = *(const f2v*)(we + 2 * HH + h0);
    f2v w3 = *(const f2v*)(we + 3 * HH + h0);
    f2v w4 = *(const f2v*)(we + 4 * HH + h0);
    f2v bb = *(const f2v*)(be + h0);
    for (int c = 0; c < CC; ++c) {
        const float* ep = edge + ((size_t)(b * CC + c) * NN + n) * EE;   // uniform addr
        float e0 = ep[0], e1 = ep[1], e2 = ep[2], e3 = ep[3], e4 = ep[4];
        float p = pres[((size_t)b * NN + n) * CC + c];
        float pre0 = bb[0] + e0 * w0[0] + e1 * w1[0] + e2 * w2[0] + e3 * w3[0] + e4 * w4[0];
        float pre1 = bb[1] + e0 * w0[1] + e1 * w1[1] + e2 * w2[1] + e3 * w3[1] + e4 * w4[1];
        h2v outv;
        outv[0] = (_Float16)(p * tanhf(pre0));
        outv[1] = (_Float16)(p * tanhf(pre1));
        *(h2v*)(pemb + (((size_t)b * NN + n) * CC + c) * HH + h0) = outv;
    }
}

// einsum l[b,n,h] = sum_c pemb[b,n,c,h] * u[b,c,h]   (cached-pemb path)
// 64 threads, each handles h pair; 32-bit pemb loads, 64-bit u loads/stores.
__global__ void k_einsum_c(const _Float16* __restrict__ pemb, const float* __restrict__ u,
                           float* __restrict__ l) {
    int bn = blockIdx.x;
    int b = bn / NN, n = bn % NN;
    int t = threadIdx.x;                      // 0..63
    const h2v* pp = (const h2v*)(pemb + (((size_t)b * NN + n) * CC) * HH) + t;
    const f2v* up = (const f2v*)(u + ((size_t)b * NN) * HH) + t;
    f2v acc = {0.0f, 0.0f};
    for (int c = 0; c < CC; ++c) {
        __builtin_prefetch(pp + (size_t)(c + 16) * (HH / 2), 0, 0);
        h2v pv = pp[(size_t)c * (HH / 2)];
        f2v uv = up[(size_t)c * (HH / 2)];
        acc[0] += (float)pv[0] * uv[0];
        acc[1] += (float)pv[1] * uv[1];
    }
    *((f2v*)(l + ((size_t)b * NN + n) * HH) + t) = acc;
}

// einsum with inline pemb recompute (fallback when ws too small for the f16 cache)
__global__ void k_einsum_r(const float* __restrict__ edge, const float* __restrict__ pres,
                           const float* __restrict__ we, const float* __restrict__ be,
                           const float* __restrict__ u, float* __restrict__ l) {
    int bn = blockIdx.x;
    int b = bn / NN, n = bn % NN;
    int t = threadIdx.x;                      // 128 threads
    float w0 = we[0 * HH + t], w1 = we[1 * HH + t], w2 = we[2 * HH + t],
          w3 = we[3 * HH + t], w4 = we[4 * HH + t], bbv = be[t];
    const float* up = u + ((size_t)b * NN) * HH + t;
    float acc = 0.0f;
    for (int c = 0; c < CC; ++c) {
        const float* ep = edge + ((size_t)(b * CC + c) * NN + n) * EE;
        float pre = bbv + ep[0] * w0 + ep[1] * w1 + ep[2] * w2 + ep[3] * w3 + ep[4] * w4;
        float p = pres[((size_t)b * NN + n) * CC + c];
        acc += p * tanhf(pre) * up[(size_t)c * HH];
    }
    l[((size_t)b * NN + n) * HH + t] = acc;
}

// ---------------------------------------------------------------------------
// WMMA GEMM: out[b, row, :] = act( A[b,row,:] @ W[128x128] + bias + (add?) )
// grid = B * NTIL blocks; block = 256 threads = 8 waves; wave w owns output
// column tile [w*16, w*16+16). K=128 -> 4 x v_wmma_f32_16x16x32_f16.
// W is transposed into LDS so both fragments load with ds_load_b128.
__global__ void k_gemm_wmma(const float* __restrict__ A, const float* __restrict__ W,
                            const float* __restrict__ bias, const float* __restrict__ add,
                            float* __restrict__ out, int relu) {
    __shared__ __align__(16) _Float16 Wt[HH * HH];   // 32 KB, Wt[col][k]
    __shared__ __align__(16) _Float16 Al[16 * HH];   // 4 KB,  Al[m][k]

    int b  = blockIdx.x / NTIL;
    int nt = blockIdx.x % NTIL;
    int tid = threadIdx.x;

    for (int i = tid; i < HH * HH; i += 256) {
        int k = i >> 7, j = i & 127;
        Wt[j * HH + k] = (_Float16)W[i];             // coalesced global read, transposed store
    }
    for (int i = tid; i < 16 * HH; i += 256) {
        int m = i >> 7, k = i & 127;
        int gr = nt * 16 + m;
        Al[i] = (gr < NN) ? (_Float16)A[((size_t)b * NN + gr) * HH + k] : (_Float16)0.0f;
    }
    __syncthreads();

    int w    = tid >> 5;
    int lane = tid & 31;
    int mrow = lane & 15;     // A-row / B-column within tile (lane % 16)
    int hilo = lane >> 4;     // lane group select

    const _Float16* arow = &Al[mrow * HH];
    const _Float16* bcol = &Wt[(w * 16 + mrow) * HH];

    v8f acc = {0.f, 0.f, 0.f, 0.f, 0.f, 0.f, 0.f, 0.f};
#pragma unroll
    for (int kk = 0; kk < HH; kk += 32) {
        // A 16x32 f16: lane holds K = kk+hilo*8 + {0..7} and {16..23}  -> 2x b128
        v8h alo = *(const v8h*)(arow + kk + hilo * 8);
        v8h ahi = *(const v8h*)(arow + kk + hilo * 8 + 16);
        v16h a = __builtin_shufflevector(alo, ahi, 0, 1, 2, 3, 4, 5, 6, 7,
                                         8, 9, 10, 11, 12, 13, 14, 15);
        // B 32x16 f16: lane holds K = kk+hilo*8 + {0..7} and {16..23} at fixed col -> 2x b128 (transposed W)
        v8h blo = *(const v8h*)(bcol + kk + hilo * 8);
        v8h bhi = *(const v8h*)(bcol + kk + hilo * 8 + 16);
        v16h bf = __builtin_shufflevector(blo, bhi, 0, 1, 2, 3, 4, 5, 6, 7,
                                          8, 9, 10, 11, 12, 13, 14, 15);
        acc = __builtin_amdgcn_wmma_f32_16x16x32_f16(false, a, false, bf,
                                                     (short)0, acc, false, false);
    }
#pragma unroll
    for (int i = 0; i < 8; ++i) {
        int m = hilo * 8 + i;              // output row within tile
        int gr = nt * 16 + m;
        int col = w * 16 + mrow;
        if (gr < NN) {
            size_t idx = ((size_t)b * NN + gr) * HH + col;
            float v = acc[i] + bias[col] + (add ? add[idx] : 0.0f);
            if (relu) v = fmaxf(v, 0.0f);
            out[idx] = v;
        }
    }
}

// ---------------------------------------------------------------------------
// Q[b] = (sum_n gamma[b,n,:] * avail[b,n]) @ wQ
__global__ void k_final(const float* __restrict__ gamma, const float* __restrict__ avail,
                        const float* __restrict__ wQ, float* __restrict__ out) {
    __shared__ float red[128];
    int b = blockIdx.x, t = threadIdx.x;
    float s = 0.0f;
    for (int n = 0; n < NN; ++n)
        s += gamma[((size_t)b * NN + n) * HH + t] * avail[b * NN + n];
    red[t] = s * wQ[t];
    __syncthreads();
    for (int off = 64; off > 0; off >>= 1) {
        if (t < off) red[t] += red[t + off];
        __syncthreads();
    }
    if (t == 0) out[b] = red[0];
}

// ---------------------------------------------------------------------------
extern "C" void kernel_launch(void* const* d_in, const int* in_sizes, int n_in,
                              void* d_out, int out_size, void* d_ws, size_t ws_size,
                              hipStream_t stream) {
    const float* ap    = (const float*)d_in[0];
    const float* ac    = (const float*)d_in[1];
    const float* x_a   = (const float*)d_in[2];
    const float* x_b   = (const float*)d_in[3];
    const float* coord = (const float*)d_in[4];
    const float* edge  = (const float*)d_in[5];
    const float* avail = (const float*)d_in[6];
    const float* w1p = (const float*)d_in[7];
    const float* b1p = (const float*)d_in[8];
    const float* w2p = (const float*)d_in[9];
    const float* b2p = (const float*)d_in[10];
    const float* wx1 = (const float*)d_in[11];
    const float* bx1 = (const float*)d_in[12];
    const float* we1 = (const float*)d_in[13];
    const float* be1 = (const float*)d_in[14];
    const float* wl1 = (const float*)d_in[15];
    const float* bl1 = (const float*)d_in[16];
    const float* wx2 = (const float*)d_in[17];
    const float* bx2 = (const float*)d_in[18];
    const float* we2 = (const float*)d_in[19];
    const float* be2 = (const float*)d_in[20];
    const float* wl2 = (const float*)d_in[21];
    const float* bl2 = (const float*)d_in[22];
    const float* wQ  = (const float*)d_in[23];

    float* ws = (float*)d_ws;
    const size_t nPres = (size_t)BB * NN * CC;     // 323,200
    const size_t nBH   = (size_t)BB * NN * HH;     // 413,696
    float* pres = ws;
    float* fx   = pres + nPres;
    float* u0   = fx + nBH;
    float* u1   = u0 + nBH;
    float* lb   = u1 + nBH;
    size_t baseBytes = (nPres + 4 * nBH) * sizeof(float);

    const size_t pembBytes = (size_t)BB * NN * CC * HH * sizeof(_Float16);  // 82.7 MB
    bool cached = ws_size >= baseBytes + pembBytes;
    _Float16* pemb = (_Float16*)((char*)d_ws + baseBytes);

    float* ubuf[2] = {u0, u1};
    const float* WE[2] = {we1, we2};
    const float* BE[2] = {be1, be2};
    const float* WL[2] = {wl1, wl2};
    const float* BL[2] = {bl1, bl2};

    // presence head + fx1
    k_presence<<<BB * CC, 128, 0, stream>>>(edge, avail, w1p, b1p, w2p, b2p, pres);
    k_fx1<<<BB * NN, 128, 0, stream>>>(ap, ac, x_a, x_b, coord, avail, wx1, bx1, fx);

    for (int r = 0; r < 2; ++r) {
        k_zero<<<(int)((nBH + 255) / 256), 256, 0, stream>>>(ubuf[0], (int)nBH);
        if (cached)
            k_pemb<<<BB * NN, 64, 0, stream>>>(edge, pres, WE[r], BE[r], pemb);
        for (int t = 0; t < TT; ++t) {
            float* uin  = ubuf[t & 1];
            float* uout = ubuf[(t + 1) & 1];
            if (cached)
                k_einsum_c<<<BB * NN, 64, 0, stream>>>(pemb, uin, lb);
            else
                k_einsum_r<<<BB * NN, 128, 0, stream>>>(edge, pres, WE[r], BE[r], uin, lb);
            k_gemm_wmma<<<BB * NTIL, 256, 0, stream>>>(lb, WL[r], BL[r], fx, uout, 1);
        }
        if (r == 0)  // fx2 = u_final @ wx2 + bx2  (u_final lives in ubuf[1] after 5 iters)
            k_gemm_wmma<<<BB * NTIL, 256, 0, stream>>>(ubuf[1], wx2, bx2, nullptr, fx, 0);
    }
    k_final<<<BB, 128, 0, stream>>>(ubuf[1], avail, wQ, (float*)d_out);
}
